// GCN_39814346834495
// MI455X (gfx1250) — matrix operand order repaired and checked by reading, not compiled, not run on hardware
//
#include <hip/hip_runtime.h>

typedef __attribute__((ext_vector_type(16))) _Float16 v16h;
typedef __attribute__((ext_vector_type(8)))  float    v8f;
typedef _Float16 half_t;

#define FH 64  // hidden width

// ---------------- utility ----------------
__global__ void k_fill(float* __restrict__ p, float v, int n) {
  int i = blockIdx.x * blockDim.x + threadIdx.x;
  if (i < n) p[i] = v;
}

// deg[dst] += ew  (deg pre-filled with 1.0 for the self-loop term)
__global__ void k_deg_edges(const int* __restrict__ dst, const float* __restrict__ ew,
                            float* __restrict__ deg, int E) {
  int e = blockIdx.x * blockDim.x + threadIdx.x;
  if (e < E) atomicAdd(&deg[dst[e]], ew[e]);
}

// deg -> dinv in place
__global__ void k_dinv(float* __restrict__ deg, int N) {
  int i = blockIdx.x * blockDim.x + threadIdx.x;
  if (i < N) {
    float d = deg[i];
    deg[i] = (d > 0.f) ? rsqrtf(d) : 0.f;
  }
}

// per-edge symmetric norm (computed once, reused by all 3 layers)
__global__ void k_norm(const int* __restrict__ src, const int* __restrict__ dst,
                       const float* __restrict__ ew, const float* __restrict__ dinv,
                       float* __restrict__ nrm, int E) {
  int e = blockIdx.x * blockDim.x + threadIdx.x;
  if (e < E) nrm[e] = dinv[src[e]] * ew[e] * dinv[dst[e]];
}

// W[K,64] f32 (row-major) -> Wt[64,K] f16 (column-major w.r.t. original)
__global__ void k_cvtW(const float* __restrict__ W, half_t* __restrict__ Wt, int K) {
  int t = blockIdx.x * blockDim.x + threadIdx.x;
  if (t >= K * FH) return;
  int k = t / FH, c = t % FH;
  Wt[(size_t)c * K + k] = (half_t)W[t];
}

// ---------------- WMMA GEMM: H[N,64] = X[N,K] @ W[K,64] ----------------
// One wave computes a full 16x64 output strip: A fragment loaded once per
// 32-wide K chunk, reused across 4 column-tile WMMAs. Weights come from the
// pre-converted f16 transposed buffer Wt[64][K]: one aligned v16h load per
// B fragment. K templated so the whole K loop unrolls.
template <int K>
__global__ void k_gemm(const float* __restrict__ X, const half_t* __restrict__ Wt,
                       float* __restrict__ H, int N, int nRowTiles) {
  const int lane = threadIdx.x;                       // 0..31
  const int tm = blockIdx.x * blockDim.y + threadIdx.y;
  if (tm >= nRowTiles) return;                        // wave-uniform

  const int half = lane >> 4;                         // 0/1
  const int sub  = lane & 15;

  // A fragment source row (clamped; garbage rows only affect unstored D rows)
  const int arow = tm * 16 + sub;
  const float* xrow = X + (size_t)((arow < N) ? arow : (N - 1)) * K;

  v8f acc[4] = {};
#pragma unroll
  for (int kb = 0; kb < K; kb += 32) {
    // A (16x32 f16): lane = row (lane&15); lanes 0-15 K=0..15,
    // lanes 16-31 K=16..31
    v16h a;
#pragma unroll
    for (int j = 0; j < 16; ++j)
      a[j] = (_Float16)xrow[kb + j + 16 * half];
#pragma unroll
    for (int t = 0; t < 4; ++t) {
      // B (32x16 f16): lane = column (lane&15); lanes 0-15 K=kb..kb+15,
      // lanes 16-31 K=kb+16..kb+31 -> 16 contiguous halves of Wt column.
      v16h b = *(const v16h*)(Wt + (size_t)(t * 16 + sub) * K + kb + 16 * half);
      acc[t] = __builtin_amdgcn_wmma_f32_16x16x32_f16(false, a, false, b,
                                                      (short)0, acc[t], false, false);
    }
  }

  // D: lane holds column sub of each tile; VGPR r holds row r + 8*half
  float* hp = H + (size_t)tm * 16 * FH + sub;
  if (tm * 16 + 16 <= N) {                            // uniform fast path
#pragma unroll
    for (int t = 0; t < 4; ++t)
#pragma unroll
      for (int r = 0; r < 8; ++r)
        hp[(size_t)(r + 8 * half) * FH + t * 16] = acc[t][r];
  } else {                                            // last partial tile only
#pragma unroll
    for (int t = 0; t < 4; ++t)
#pragma unroll
      for (int r = 0; r < 8; ++r) {
        int orow = tm * 16 + r + 8 * half;
        if (orow < N) hp[(size_t)(r + 8 * half) * FH + t * 16] = acc[t][r];
      }
  }
}

// Runtime-K fallback (only used if the input width is not 128/64). K % 32 == 0.
__global__ void k_gemm_dyn(const float* __restrict__ X, const half_t* __restrict__ Wt,
                           float* __restrict__ H, int N, int K, int nRowTiles) {
  const int lane = threadIdx.x;
  const int tm = blockIdx.x * blockDim.y + threadIdx.y;
  if (tm >= nRowTiles) return;
  const int half = lane >> 4, sub = lane & 15;
  const int arow = tm * 16 + sub;
  const float* xrow = X + (size_t)((arow < N) ? arow : (N - 1)) * K;
  v8f acc[4] = {};
  for (int kb = 0; kb < K; kb += 32) {
    v16h a;
#pragma unroll
    for (int j = 0; j < 16; ++j)
      a[j] = (_Float16)xrow[kb + j + 16 * half];
#pragma unroll
    for (int t = 0; t < 4; ++t) {
      v16h b = *(const v16h*)(Wt + (size_t)(t * 16 + sub) * K + kb + 16 * half);
      acc[t] = __builtin_amdgcn_wmma_f32_16x16x32_f16(false, a, false, b,
                                                      (short)0, acc[t], false, false);
    }
  }
  float* hp = H + (size_t)tm * 16 * FH + sub;
#pragma unroll
  for (int t = 0; t < 4; ++t)
#pragma unroll
    for (int r = 0; r < 8; ++r) {
      int orow = tm * 16 + r + 8 * half;
      if (orow < N) hp[(size_t)(r + 8 * half) * FH + t * 16] = acc[t][r];
    }
}

// OUT[i,:] = dinv[i]^2 * H[i,:] + bias   (self-loop edge + bias; 16 lanes/node)
__global__ void k_agg_init(const float* __restrict__ H, const float* __restrict__ dinv,
                           const float* __restrict__ bias, float* __restrict__ OUT, int N) {
  int t = blockIdx.x * blockDim.x + threadIdx.x;
  int i = t >> 4, f4 = t & 15;
  if (i >= N) return;
  float di = dinv[i];
  float w = di * di;
  float4 h = ((const float4*)(H + (size_t)i * FH))[f4];
  float4 b = ((const float4*)bias)[f4];
  float4 o = { fmaf(w, h.x, b.x), fmaf(w, h.y, b.y),
               fmaf(w, h.z, b.z), fmaf(w, h.w, b.w) };
  ((float4*)(OUT + (size_t)i * FH))[f4] = o;
}

// OUT[dst,:] += norm_e * H[src,:]   (16 lanes per edge, float4 gather, 4 atomics)
__global__ void k_agg_edges(const int* __restrict__ src, const int* __restrict__ dst,
                            const float* __restrict__ nrm, const float* __restrict__ H,
                            float* __restrict__ OUT, int E) {
  int t = blockIdx.x * blockDim.x + threadIdx.x;
  int e = t >> 4, f4 = t & 15;
  if (e >= E) return;
  int s = src[e], d = dst[e];
  float w = nrm[e];
  float4 h = ((const float4*)(H + (size_t)s * FH))[f4];
  float* o = OUT + (size_t)d * FH + (size_t)f4 * 4;
  atomicAdd(o + 0, w * h.x);
  atomicAdd(o + 1, w * h.y);
  atomicAdd(o + 2, w * h.z);
  atomicAdd(o + 3, w * h.w);
}

__global__ void k_relu(float* __restrict__ p, int N) {
  int t = blockIdx.x * blockDim.x + threadIdx.x;   // N*16 threads
  if (t >= N * 16) return;
  float4 v = ((float4*)p)[t];
  v.x = fmaxf(v.x, 0.f); v.y = fmaxf(v.y, 0.f);
  v.z = fmaxf(v.z, 0.f); v.w = fmaxf(v.w, 0.f);
  ((float4*)p)[t] = v;
}

// pooled[batch[i],:] += B[i,:]; cnt[batch[i]] += 1
__global__ void k_pool(const float* __restrict__ Bm, const int* __restrict__ batch,
                       float* __restrict__ pooled, float* __restrict__ cnt, int N) {
  int t = blockIdx.x * blockDim.x + threadIdx.x;
  int i = t >> 4, f4 = t & 15;
  if (i >= N) return;
  int g = batch[i];
  float4 v = ((const float4*)(Bm + (size_t)i * FH))[f4];
  float* p = pooled + (size_t)g * FH + (size_t)f4 * 4;
  atomicAdd(p + 0, v.x);
  atomicAdd(p + 1, v.y);
  atomicAdd(p + 2, v.z);
  atomicAdd(p + 3, v.w);
  if (f4 == 0) atomicAdd(&cnt[g], 1.f);
}

// out[g,c] = (pooled[g,:]/max(cnt,1)) @ Wl[:,c] + bl[c]
__global__ void k_final(const float* __restrict__ pooled, const float* __restrict__ cnt,
                        const float* __restrict__ Wl, const float* __restrict__ bl,
                        float* __restrict__ out, int G, int C) {
  int t = blockIdx.x * blockDim.x + threadIdx.x;
  if (t >= G * C) return;
  int g = t / C, c = t % C;
  float inv = 1.f / fmaxf(cnt[g], 1.f);
  float acc = bl[c];
#pragma unroll 8
  for (int f = 0; f < FH; ++f)
    acc = fmaf(pooled[(size_t)g * FH + f] * inv, Wl[f * C + c], acc);
  out[t] = acc;
}

// ---------------- host ----------------
static inline void launch_gemm(const float* X, const half_t* Wt, float* H,
                               int N, int K, int rowTiles, hipStream_t stream) {
  const dim3 blk(32, 4);
  const unsigned nb = (unsigned)((rowTiles + 3) / 4);
  if (K == 128)      k_gemm<128><<<nb, blk, 0, stream>>>(X, Wt, H, N, rowTiles);
  else if (K == 64)  k_gemm<64><<<nb, blk, 0, stream>>>(X, Wt, H, N, rowTiles);
  else               k_gemm_dyn<<<nb, blk, 0, stream>>>(X, Wt, H, N, K, rowTiles);
}

extern "C" void kernel_launch(void* const* d_in, const int* in_sizes, int n_in,
                              void* d_out, int out_size, void* d_ws, size_t ws_size,
                              hipStream_t stream) {
  const float* x     = (const float*)d_in[0];
  const int*   ei    = (const int*)d_in[1];
  const float* ea    = (const float*)d_in[2];
  const int*   batch = (const int*)d_in[3];
  const float* W1 = (const float*)d_in[4];  const float* b1 = (const float*)d_in[5];
  const float* W2 = (const float*)d_in[6];  const float* b2 = (const float*)d_in[7];
  const float* W3 = (const float*)d_in[8];  const float* b3 = (const float*)d_in[9];
  const float* Wl = (const float*)d_in[10]; const float* bl = (const float*)d_in[11];

  const int C = in_sizes[11];            // 10 classes
  const int N = in_sizes[3];             // nodes (batch length)
  const int E = in_sizes[1] / 2;         // edges
  const int G = out_size / C;            // graphs
  const int K1 = in_sizes[0] / N;        // 128
  const int* src = ei;
  const int* dst = ei + E;

  // workspace layout (floats, 256-elem = 1KB aligned)
  auto al = [](size_t v) { return (v + 255) & ~(size_t)255; };
  float* ws     = (float*)d_ws;
  float* dinv   = ws;
  float* nrm    = dinv + al((size_t)N);
  float* A      = nrm  + al((size_t)E);
  float* Bm     = A    + al((size_t)N * FH);
  float* pooled = Bm   + al((size_t)N * FH);
  float* cnt    = pooled + (size_t)G * FH;
  half_t* Wt    = (half_t*)(pooled + al((size_t)G * FH + (size_t)G)); // f16 weights

  const int T = 256;
  auto gr = [&](long long n) { return (unsigned)((n + T - 1) / T); };

  // degree / norm (once for all layers)
  k_fill<<<gr(N), T, 0, stream>>>(dinv, 1.0f, N);
  k_deg_edges<<<gr(E), T, 0, stream>>>(dst, ea, dinv, E);
  k_dinv<<<gr(N), T, 0, stream>>>(dinv, N);
  k_norm<<<gr(E), T, 0, stream>>>(src, dst, ea, dinv, nrm, E);

  const int rowTiles = (N + 15) / 16;
  const long long nt = (long long)N * 16;
  const long long et = (long long)E * 16;

  // layer 1
  k_cvtW<<<gr((long long)K1 * FH), T, 0, stream>>>(W1, Wt, K1);
  launch_gemm(x, Wt, A, N, K1, rowTiles, stream);
  k_agg_init<<<gr(nt), T, 0, stream>>>(A, dinv, b1, Bm, N);
  k_agg_edges<<<gr(et), T, 0, stream>>>(src, dst, nrm, A, Bm, E);
  k_relu<<<gr(nt), T, 0, stream>>>(Bm, N);
  // layer 2
  k_cvtW<<<gr((long long)FH * FH), T, 0, stream>>>(W2, Wt, FH);
  launch_gemm(Bm, Wt, A, N, FH, rowTiles, stream);
  k_agg_init<<<gr(nt), T, 0, stream>>>(A, dinv, b2, Bm, N);
  k_agg_edges<<<gr(et), T, 0, stream>>>(src, dst, nrm, A, Bm, E);
  k_relu<<<gr(nt), T, 0, stream>>>(Bm, N);
  // layer 3
  k_cvtW<<<gr((long long)FH * FH), T, 0, stream>>>(W3, Wt, FH);
  launch_gemm(Bm, Wt, A, N, FH, rowTiles, stream);
  k_agg_init<<<gr(nt), T, 0, stream>>>(A, dinv, b3, Bm, N);
  k_agg_edges<<<gr(et), T, 0, stream>>>(src, dst, nrm, A, Bm, E);
  k_relu<<<gr(nt), T, 0, stream>>>(Bm, N);

  // pool + classifier
  k_fill<<<gr((long long)G * (FH + 1)), T, 0, stream>>>(pooled, 0.f, G * (FH + 1));
  k_pool<<<gr(nt), T, 0, stream>>>(Bm, batch, pooled, cnt, N);
  k_final<<<gr((long long)G * C), T, 0, stream>>>(pooled, cnt, Wl, bl, (float*)d_out, G, C);
}
